// Int8PhiMLP_61005715473017
// MI455X (gfx1250) — 8-bit matrix operand order repaired and checked by reading, compiled, not run on hardware
//
#include <hip/hip_runtime.h>
#include <stdint.h>


typedef int v8i __attribute__((ext_vector_type(8)));

#define M_DIM 4096
#define H_DIM 2560
#define I_DIM 10240

#define BLK_M 128
#define BLK_N 128
#define KB    64
#define LDS_STRIDE 80
#define SLAB  (BLK_M * LDS_STRIDE)

union Frag {
    v8i  v;
    int2 d2[4];
    int4 d4[2];
};

__device__ __forceinline__ void async_ld16(uint32_t lds_addr, const int8_t* gptr)
{
    asm volatile("global_load_async_to_lds_b128 %0, %1, off"
                 :: "v"(lds_addr), "v"(gptr)
                 : "memory");
}

__global__ __launch_bounds__(256) void quant_token_kernel(
    const float* __restrict__ x, int8_t* __restrict__ q, float* __restrict__ s_out)
{
    __shared__ float red[256];
    const int m   = blockIdx.x;
    const int tid = threadIdx.x;
    const float* xr = x + (size_t)m * H_DIM;

    float amax = 0.0f;
#pragma unroll
    for (int i = 0; i < 10; ++i)
        amax = fmaxf(amax, fabsf(xr[tid + i * 256]));
    red[tid] = amax;
    __syncthreads();
#pragma unroll
    for (int off = 128; off > 0; off >>= 1) {
        if (tid < off) red[tid] = fmaxf(red[tid], red[tid + off]);
        __syncthreads();
    }
    const float s   = fmaxf(red[0] * (1.0f / 127.0f), 1e-8f);
    const float inv = 1.0f / s;
    if (tid == 0) s_out[m] = s;

    int8_t* qr = q + (size_t)m * H_DIM;
#pragma unroll
    for (int i = 0; i < 10; ++i) {
        float r = rintf(xr[tid + i * 256] * inv);
        r = fminf(fmaxf(r, -128.0f), 127.0f);
        qr[tid + i * 256] = (int8_t)r;
    }
}

__device__ __forceinline__ float gelu_new(float u)
{
    const float c = 0.7978845608028654f;
    return 0.5f * u * (1.0f + tanhf(c * (u + 0.044715f * u * u * u)));
}

template <int N_DIM, int K_DIM, int MODE>
__global__ __launch_bounds__(256) void gemm_i8_wmma_kernel(
    const int8_t* __restrict__ A,
    const int8_t* __restrict__ W,
    const float*  __restrict__ rowScale,
    const float*  __restrict__ bias,
    const float*  __restrict__ sc0_p,
    const float*  __restrict__ sc1_p,
    int8_t* __restrict__ out8,
    float*  __restrict__ outf)
{
    static_assert(K_DIM >= 2 * KB, "pipeline needs at least two K slabs");

    __shared__ __align__(16) int8_t lds[2][2][SLAB];

    const int tid  = threadIdx.x;
    const int lane = tid & 31;
    const int wave = tid >> 5;
    const int hl   = lane >> 4;
    const int l16  = lane & 15;
    const int waveM = (wave >> 2) * 64;
    const int waveN = (wave & 3) * 32;

    const int m0 = blockIdx.x * BLK_M;
    const int n0 = blockIdx.y * BLK_N;

    const int r0 = tid >> 2;
    const int ko = (tid & 3) * 16;
    const int8_t* gA = A + (size_t)(m0 + r0) * K_DIM + ko;
    const int8_t* gW = W + (size_t)(n0 + r0) * K_DIM + ko;
    const size_t rowskip = (size_t)64 * K_DIM;
    const uint32_t lofs = (uint32_t)(r0 * LDS_STRIDE + ko);

    v8i acc[4][2];
#pragma unroll
    for (int i = 0; i < 4; ++i)
#pragma unroll
        for (int j = 0; j < 2; ++j)
            acc[i][j] = (v8i){0, 0, 0, 0, 0, 0, 0, 0};

    auto stage_async = [&](int buf, int kb) {
        const uint32_t la = (uint32_t)(size_t)(&lds[buf][0][0]) + lofs;
        const uint32_t lw = (uint32_t)(size_t)(&lds[buf][1][0]) + lofs;
        async_ld16(la,                             gA + kb);
        async_ld16(la + 64u * LDS_STRIDE,          gA + kb + rowskip);
        async_ld16(lw,                             gW + kb);
        async_ld16(lw + 64u * LDS_STRIDE,          gW + kb + rowskip);
    };

    stage_async(0, 0);
    stage_async(1, KB);
    asm volatile("s_wait_asynccnt 0x4" ::: "memory");
    __syncthreads();

    int cur = 0;
    for (int kb = 0; kb < K_DIM; kb += KB) {
        const int8_t* la = &lds[cur][0][0];
        const int8_t* lw = &lds[cur][1][0];

        Frag bf0, bf1;
        {
            const int8_t* rw = lw + (waveN + l16) * LDS_STRIDE + 16 * hl;
            bf0.d4[0] = *(const int4*)(rw);
            bf0.d4[1] = *(const int4*)(rw + 32);
            const int8_t* rw1 = rw + 16 * LDS_STRIDE;
            bf1.d4[0] = *(const int4*)(rw1);
            bf1.d4[1] = *(const int4*)(rw1 + 32);
        }

#pragma unroll
        for (int i = 0; i < 4; ++i) {
            Frag af;
            const int8_t* ra = la + (waveM + i * 16 + l16) * LDS_STRIDE + hl * 16;
            af.d4[0] = *(const int4*)(ra);
            af.d4[1] = *(const int4*)(ra + 32);
            acc[i][0] = __builtin_amdgcn_wmma_i32_16x16x64_iu8(
                true, af.v, true, bf0.v, acc[i][0], false, false);
            acc[i][1] = __builtin_amdgcn_wmma_i32_16x16x64_iu8(
                true, af.v, true, bf1.v, acc[i][1], false, false);
        }

        if (kb + KB < K_DIM)
            asm volatile("s_wait_asynccnt 0x0" ::: "memory");
        __syncthreads();
        if (kb + 2 * KB < K_DIM)
            stage_async(cur, kb + 2 * KB);
        cur ^= 1;
    }

    const float sc0 = *sc0_p;
    const float sc1 = *sc1_p;
#pragma unroll
    for (int i = 0; i < 4; ++i) {
#pragma unroll
        for (int j = 0; j < 2; ++j) {
            const int ng = n0 + waveN + j * 16 + l16;
            const float bn = bias[ng];
#pragma unroll
            for (int r = 0; r < 8; ++r) {
                const int   mg = m0 + waveM + i * 16 + r + 8 * hl;
                const float av = (float)acc[i][j][r];
                if (MODE == 0) {
                    const float h  = av * (rowScale[mg] * sc0) + bn;
                    const float g  = gelu_new(h);
                    float qv = rintf(g * (1.0f / sc1));
                    qv = fminf(fmaxf(qv, -128.0f), 127.0f);
                    out8[(size_t)mg * N_DIM + ng] = (int8_t)qv;
                } else {
                    outf[(size_t)mg * N_DIM + ng] = av * (sc0 * sc1) + bn;
                }
            }
        }
    }
}

extern "C" void kernel_launch(void* const* d_in, const int* in_sizes, int n_in,
                              void* d_out, int out_size, void* d_ws, size_t ws_size,
                              hipStream_t stream)
{
    (void)in_sizes; (void)n_in; (void)out_size; (void)ws_size;

    const float*  x           = (const float*)d_in[0];
    const int8_t* w1_q        = (const int8_t*)d_in[1];
    const float*  w1_scale    = (const float*)d_in[2];
    const float*  b1          = (const float*)d_in[3];
    const int8_t* w2_q        = (const int8_t*)d_in[4];
    const float*  w2_scale    = (const float*)d_in[5];
    const float*  b2          = (const float*)d_in[6];
    const float*  quant_scale = (const float*)d_in[7];
    float* out = (float*)d_out;

    int8_t* q1 = (int8_t*)d_ws;
    float*  s1 = (float*)((char*)d_ws + (size_t)M_DIM * H_DIM);
    int8_t* q2 = (int8_t*)((char*)d_ws + (size_t)M_DIM * H_DIM
                                       + (size_t)M_DIM * sizeof(float));

    quant_token_kernel<<<M_DIM, 256, 0, stream>>>(x, q1, s1);

    {
        dim3 grid(M_DIM / BLK_M, I_DIM / BLK_N);
        gemm_i8_wmma_kernel<I_DIM, H_DIM, 0><<<grid, 256, 0, stream>>>(
            q1, w1_q, s1, b1, w1_scale, quant_scale, q2, nullptr);
    }

    {
        dim3 grid(M_DIM / BLK_M, H_DIM / BLK_N);
        gemm_i8_wmma_kernel<H_DIM, I_DIM, 1><<<grid, 256, 0, stream>>>(
            q2, w2_q, nullptr, b2, quant_scale, w2_scale, nullptr, out);
    }
}
